// TreeTransformerClassifier_66133906424370
// MI455X (gfx1250) — matrix operand order repaired and checked by reading, not compiled, not run on hardware
//
#include <hip/hip_runtime.h>
#include <hip/hip_fp16.h>

// ---------------------------------------------------------------------------
// TreeTransformer classifier for gfx1250 (MI455X).
// GEMMs use v_wmma_f32_16x16x32_f16 (f16 A/B, f32 accumulate).
// Attention staging uses global_load_async_to_lds_b128 (ASYNCcnt path).
// ---------------------------------------------------------------------------

typedef __attribute__((ext_vector_type(16))) _Float16 v16h;
typedef __attribute__((ext_vector_type(8)))  float    v8f;

#define DEVFN static __device__ __forceinline__

DEVFN float wave_sum(float v) {
#pragma unroll
  for (int o = 16; o; o >>= 1) v += __shfl_xor(v, o, 32);
  return v;
}

DEVFN float gelu_exact(float x) {
  return 0.5f * x * (1.0f + erff(x * 0.7071067811865475f));
}

// ---------------------------------------------------------------------------
// Generic WMMA GEMM:  Cout[M,N] = op(A[M,K] @ W[K,N] + bias) (+ res)
//   W supplied pre-transposed + converted:  Wt[N][K] in f16.
//   act==1 -> exact GELU applied after bias.
// Block = 256 threads = 8 waves (4 row-blocks x 2 col-blocks).
// Block tile 64x128, wave tile 16x64 (four 16x16 f32 accumulators), K step 32.
// One A fragment feeds 4 WMMAs -> A(fp32) traffic amortized 4x.
// M must be a multiple of 16 (true for every call here); N multiple of 128.
// ---------------------------------------------------------------------------
__global__ __launch_bounds__(256)
void wmma_gemm(const float* __restrict__ A, const _Float16* __restrict__ Wt,
               const float* __restrict__ bias, const float* __restrict__ res,
               float* __restrict__ Cout, int M, int Ncols, int K, int act) {
  const int wave = threadIdx.x >> 5;
  const int lane = threadIdx.x & 31;
  const int wr = wave & 3;
  const int wc = wave >> 2;
  const int mbase = blockIdx.x * 64 + wr * 16;
  const int nbase = blockIdx.y * 128 + wc * 64;
  if (mbase >= M) return;  // wave-uniform: EXEC stays all-1 for waves that
                           // reach the WMMA ops.
  const int g   = lane >> 4;   // lane-group (0/1) per ISA fragment layout
  const int l15 = lane & 15;
  const int mrow = mbase + l15;          // A fragment row for this lane

  v8f acc[4];
#pragma unroll
  for (int f = 0; f < 4; ++f) acc[f] = (v8f){0.f,0.f,0.f,0.f,0.f,0.f,0.f,0.f};

  const float* arow = A + (size_t)mrow * K;
  const _Float16* bp[4];
#pragma unroll
  for (int f = 0; f < 4; ++f)
    bp[f] = Wt + (size_t)(nbase + 16 * f + l15) * K + 16 * g;

  for (int kb = 0; kb < K; kb += 32) {
    // A 16x32 f16 fragment: lane-group g takes K = kb+16g..+15 contiguous
    // (elems 0..15), matching B.
    const float4* a4 = (const float4*)(arow + kb + 16 * g);
    float4 p0 = a4[0], p1 = a4[1], p2 = a4[2], p3 = a4[3];
    v16h a;
    a[0]  = (_Float16)p0.x; a[1]  = (_Float16)p0.y;
    a[2]  = (_Float16)p0.z; a[3]  = (_Float16)p0.w;
    a[4]  = (_Float16)p1.x; a[5]  = (_Float16)p1.y;
    a[6]  = (_Float16)p1.z; a[7]  = (_Float16)p1.w;
    a[8]  = (_Float16)p2.x; a[9]  = (_Float16)p2.y;
    a[10] = (_Float16)p2.z; a[11] = (_Float16)p2.w;
    a[12] = (_Float16)p3.x; a[13] = (_Float16)p3.y;
    a[14] = (_Float16)p3.z; a[15] = (_Float16)p3.w;

    if (kb + 32 < K) {  // gfx1250 global_prefetch_b8 on the weight stream
      __builtin_prefetch((const void*)(bp[0] + kb + 32), 0, 3);
      __builtin_prefetch((const void*)(bp[2] + kb + 32), 0, 3);
    }

    // B 32x16 f16 fragments: lane = column n, K = kb+16g..+15 contiguous.
#pragma unroll
    for (int f = 0; f < 4; ++f) {
      v16h b = *(const v16h*)(bp[f] + kb);
      acc[f] = __builtin_amdgcn_wmma_f32_16x16x32_f16(false, a, false, b,
                                                      (short)0, acc[f],
                                                      false, false);
    }
  }

  // C/D layout: VGPR j -> row mbase + 8g + j, lane&15 -> column.
#pragma unroll
  for (int j = 0; j < 8; ++j) {
    const int m = mbase + 8 * g + j;
    const size_t base = (size_t)m * Ncols;
#pragma unroll
    for (int f = 0; f < 4; ++f) {
      const int n = nbase + 16 * f + l15;
      float v0 = acc[f][j] + bias[n];
      if (act) v0 = gelu_exact(v0);
      if (res) v0 += res[base + n];
      Cout[base + n] = v0;
    }
  }
}

// ---------------------------------------------------------------------------
// Utility / small kernels
// ---------------------------------------------------------------------------

// Sinusoidal PE for seq=8, d=256.
__global__ void pe_init(float* __restrict__ pe) {
  const int c = blockIdx.x, d = threadIdx.x;
  const int i2 = d & ~1;
  const float div = expf(-(float)i2 * (logf(10000.0f) / 256.0f));
  const float ang = (float)c * div;
  pe[c * 256 + d] = (d & 1) ? cosf(ang) : sinf(ang);
}

// Wt[n*K + k] = (f16) W[k*N + n]
__global__ void cvt_wt(const float* __restrict__ W, _Float16* __restrict__ Wt,
                       int K, int Ncols) {
  int idx = blockIdx.x * 256 + threadIdx.x;
  if (idx >= K * Ncols) return;
  int k = idx / Ncols, n = idx % Ncols;
  Wt[(size_t)n * K + k] = (_Float16)W[idx];
}

// hall[row] = emb[types[row]]
__global__ void embed_gather(const int* __restrict__ types,
                             const float* __restrict__ emb,
                             float* __restrict__ hall) {
  const int row = blockIdx.x, d = threadIdx.x;
  const int t = types[row];
  hall[(size_t)row * 256 + d] = emb[(size_t)t * 256 + d];
}

// x[r] = children(hall) + pe[c]   (r = local child row in chunk)
__global__ void gather_children_pe(const float* __restrict__ hall,
                                   const float* __restrict__ pe,
                                   float* __restrict__ x,
                                   int off_child, int P, int g0) {
  const int r = blockIdx.x, d = threadIdx.x;
  const int gidx = g0 + (r >> 3), c = r & 7;
  const int b = gidx / P, p = gidx % P;
  const float* src = hall + ((size_t)b * 4681 + off_child + p * 8 + c) * 256;
  x[(size_t)r * 256 + d] = src[d] + pe[c * 256 + d];
}

// par[i] = hall[parent node of group g0+i]
__global__ void gather_rows(const float* __restrict__ hall,
                            float* __restrict__ out,
                            int g0, int P, int off_l) {
  const int i = blockIdx.x, d = threadIdx.x;
  const int gidx = g0 + i;
  const int b = gidx / P, p = gidx % P;
  out[(size_t)i * 256 + d] = hall[((size_t)b * 4681 + off_l + p) * 256 + d];
}

// Sibling attention: one block per group of 8 children. 8 heads, DK=32.
// q/k/v rows staged to LDS with gfx1250 async global->LDS DMA (ASYNCcnt).
__global__ __launch_bounds__(256)
void sib_attn(const float* __restrict__ q, const float* __restrict__ k,
              const float* __restrict__ v, float* __restrict__ attn) {
  __shared__ float smem[3 * 8 * 256];          // sq | sk | sv, 24KB contiguous
  __shared__ float sc[8][8][8];                // [head][i][j]
  float* sq = smem;
  float* sk = smem + 2048;
  float* sv = smem + 4096;
  const size_t r0 = (size_t)blockIdx.x * 8;

  {
    const float* gsrc[3] = {q + r0 * 256, k + r0 * 256, v + r0 * 256};
#pragma unroll
    for (int part = 0; part < 3; ++part) {
      const char* gbase = (const char*)gsrc[part];
      // generic->LDS: aperture layout keeps the LDS byte offset in addr[31:0]
      unsigned lbase = (unsigned)(size_t)(smem + part * 2048);
#pragma unroll
      for (int it = 0; it < 2; ++it) {
        const int idx = threadIdx.x + it * 256;   // 512 x 16B = 8KB per part
        unsigned loff = lbase + idx * 16;
        const char* gp = gbase + (size_t)idx * 16;
        asm volatile("global_load_async_to_lds_b128 %0, %1, off"
                     :: "v"(loff), "v"(gp) : "memory");
      }
    }
    asm volatile("s_wait_asynccnt 0x0" ::: "memory");
  }
  __syncthreads();

  const float scale = 0.17677669529663687f;  // 1/sqrt(32)
  for (int t = threadIdx.x; t < 512; t += 256) {
    const int h = t >> 6, i = (t >> 3) & 7, j = t & 7;
    float s = 0.f;
#pragma unroll
    for (int d = 0; d < 32; ++d)
      s += sq[i * 256 + h * 32 + d] * sk[j * 256 + h * 32 + d];
    sc[h][i][j] = s * scale;
  }
  __syncthreads();
  if (threadIdx.x < 64) {
    const int h = threadIdx.x >> 3, i = threadIdx.x & 7;
    float mx = -1e30f;
#pragma unroll
    for (int j = 0; j < 8; ++j) mx = fmaxf(mx, sc[h][i][j]);
    float sum = 0.f;
#pragma unroll
    for (int j = 0; j < 8; ++j) { float e = expf(sc[h][i][j] - mx); sc[h][i][j] = e; sum += e; }
    const float inv = 1.f / sum;
#pragma unroll
    for (int j = 0; j < 8; ++j) sc[h][i][j] *= inv;
  }
  __syncthreads();
  for (int t = threadIdx.x; t < 2048; t += 256) {
    const int i = t >> 8, d = t & 255, h = d >> 5;
    float o = 0.f;
#pragma unroll
    for (int j = 0; j < 8; ++j) o += sc[h][i][j] * sv[j * 256 + d];
    attn[(r0 + i) * 256 + d] = o;
  }
}

// Parent attention: pq[G,256] vs nk/nv[G*8,256]; one wave per group.
__global__ __launch_bounds__(256)
void par_attn(const float* __restrict__ pq, const float* __restrict__ nk,
              const float* __restrict__ nv, float* __restrict__ ctx, int G) {
  const int gidx = blockIdx.x * 8 + (threadIdx.x >> 5);
  const int lane = threadIdx.x & 31;
  if (gidx >= G) return;
  const float scale = 0.17677669529663687f;
  const float* pqr = pq + (size_t)gidx * 256;
  const float* nkr = nk + (size_t)gidx * 8 * 256;
  const float* nvr = nv + (size_t)gidx * 8 * 256;
  for (int h = 0; h < 8; ++h) {
    const float qd = pqr[h * 32 + lane];
    float s[8];
#pragma unroll
    for (int j = 0; j < 8; ++j)
      s[j] = wave_sum(qd * nkr[j * 256 + h * 32 + lane]) * scale;
    float mx = -1e30f;
#pragma unroll
    for (int j = 0; j < 8; ++j) mx = fmaxf(mx, s[j]);
    float sum = 0.f;
#pragma unroll
    for (int j = 0; j < 8; ++j) { s[j] = expf(s[j] - mx); sum += s[j]; }
    const float inv = 1.f / sum;
    float o = 0.f;
#pragma unroll
    for (int j = 0; j < 8; ++j) o += s[j] * inv * nvr[j * 256 + h * 32 + lane];
    ctx[(size_t)gidx * 256 + h * 32 + lane] = o;
  }
}

// Row LayerNorm over 256 dims, wave per row.  dense in->out (may alias).
__global__ __launch_bounds__(256)
void ln_dense(const float* __restrict__ in, float* __restrict__ out,
              const float* __restrict__ ga, const float* __restrict__ be,
              int rows) {
  const int r = blockIdx.x * 8 + (threadIdx.x >> 5);
  const int lane = threadIdx.x & 31;
  if (r >= rows) return;
  const float* x = in + (size_t)r * 256;
  float vals[8]; float s = 0.f;
#pragma unroll
  for (int t = 0; t < 8; ++t) { vals[t] = x[t * 32 + lane]; s += vals[t]; }
  const float mean = wave_sum(s) * (1.f / 256.f);
  float s2 = 0.f;
#pragma unroll
  for (int t = 0; t < 8; ++t) { float d = vals[t] - mean; s2 += d * d; }
  const float inv = rsqrtf(wave_sum(s2) * (1.f / 256.f) + 1e-5f);
  float* o = out + (size_t)r * 256;
#pragma unroll
  for (int t = 0; t < 8; ++t) {
    const int d = t * 32 + lane;
    o[d] = (vals[t] - mean) * inv * ga[d] + be[d];
  }
}

// LN + scatter into hall:  local row i -> g=base0+i, b=g/PS, node=off_l+g%PS.
__global__ __launch_bounds__(256)
void ln_scatter(const float* __restrict__ in, float* __restrict__ hall,
                const float* __restrict__ ga, const float* __restrict__ be,
                int rows, int base0, int PS, int off_l) {
  const int r = blockIdx.x * 8 + (threadIdx.x >> 5);
  const int lane = threadIdx.x & 31;
  if (r >= rows) return;
  const float* x = in + (size_t)r * 256;
  float vals[8]; float s = 0.f;
#pragma unroll
  for (int t = 0; t < 8; ++t) { vals[t] = x[t * 32 + lane]; s += vals[t]; }
  const float mean = wave_sum(s) * (1.f / 256.f);
  float s2 = 0.f;
#pragma unroll
  for (int t = 0; t < 8; ++t) { float d = vals[t] - mean; s2 += d * d; }
  const float inv = rsqrtf(wave_sum(s2) * (1.f / 256.f) + 1e-5f);
  const int gidx = base0 + r;
  const int b = gidx / PS, p = gidx % PS;
  float* o = hall + ((size_t)b * 4681 + off_l + p) * 256;
#pragma unroll
  for (int t = 0; t < 8; ++t) {
    const int d = t * 32 + lane;
    o[d] = (vals[t] - mean) * inv * ga[d] + be[d];
  }
}

// Top-down combine: ctx[r] = LN(hall[parent] + hall[child]), wave per row.
__global__ __launch_bounds__(256)
void td_combine(const float* __restrict__ hall, float* __restrict__ ctx,
                int rows, int r0, int S, int off_par, int off_l,
                const float* __restrict__ ga, const float* __restrict__ be) {
  const int r = blockIdx.x * 8 + (threadIdx.x >> 5);
  const int lane = threadIdx.x & 31;
  if (r >= rows) return;
  const int rg = r0 + r;
  const int b = rg / S, idx = rg % S, pidx = idx >> 3;
  const float* xp = hall + ((size_t)b * 4681 + off_par + pidx) * 256;
  const float* xc = hall + ((size_t)b * 4681 + off_l + idx) * 256;
  float vals[8]; float s = 0.f;
#pragma unroll
  for (int t = 0; t < 8; ++t) {
    const int d = t * 32 + lane;
    vals[t] = xp[d] + xc[d]; s += vals[t];
  }
  const float mean = wave_sum(s) * (1.f / 256.f);
  float s2 = 0.f;
#pragma unroll
  for (int t = 0; t < 8; ++t) { float d = vals[t] - mean; s2 += d * d; }
  const float inv = rsqrtf(wave_sum(s2) * (1.f / 256.f) + 1e-5f);
  float* o = ctx + (size_t)r * 256;
#pragma unroll
  for (int t = 0; t < 8; ++t) {
    const int d = t * 32 + lane;
    o[d] = (vals[t] - mean) * inv * ga[d] + be[d];
  }
}

// Gate scores: s[row] = hall[row] . gate_w + gate_b, wave per row.
__global__ __launch_bounds__(256)
void gate_rows(const float* __restrict__ hall, const float* __restrict__ gw,
               const float* __restrict__ gb, float* __restrict__ s, int rows) {
  const int r = blockIdx.x * 8 + (threadIdx.x >> 5);
  const int lane = threadIdx.x & 31;
  if (r >= rows) return;
  const float* x = hall + (size_t)r * 256;
  float acc = 0.f;
#pragma unroll
  for (int t = 0; t < 8; ++t) acc += x[t * 32 + lane] * gw[t * 32 + lane];
  acc = wave_sum(acc);
  if (lane == 0) s[r] = acc + gb[0];
}

// Softmax over N=4681 nodes per tree, one block per tree, in place.
__global__ __launch_bounds__(256)
void softmax_rows(float* __restrict__ s, int Nn) {
  __shared__ float red[256];
  float* row = s + (size_t)blockIdx.x * Nn;
  float mx = -1e30f;
  for (int n = threadIdx.x; n < Nn; n += 256) mx = fmaxf(mx, row[n]);
  red[threadIdx.x] = mx; __syncthreads();
  for (int o = 128; o; o >>= 1) {
    if (threadIdx.x < o) red[threadIdx.x] = fmaxf(red[threadIdx.x], red[threadIdx.x + o]);
    __syncthreads();
  }
  mx = red[0]; __syncthreads();
  float sum = 0.f;
  for (int n = threadIdx.x; n < Nn; n += 256) {
    float e = expf(row[n] - mx); row[n] = e; sum += e;
  }
  red[threadIdx.x] = sum; __syncthreads();
  for (int o = 128; o; o >>= 1) {
    if (threadIdx.x < o) red[threadIdx.x] += red[threadIdx.x + o];
    __syncthreads();
  }
  const float inv = 1.f / red[0];
  for (int n = threadIdx.x; n < Nn; n += 256) row[n] *= inv;
}

// pooled[b,d] = sum_n a[b,n] * hall[b,n,d] ; one block per tree, thread per d.
__global__ __launch_bounds__(256)
void pool_rows(const float* __restrict__ a, const float* __restrict__ hall,
               float* __restrict__ pooled, int Nn) {
  const int b = blockIdx.x, d = threadIdx.x;
  const float* ab = a + (size_t)b * Nn;
  const float* hb = hall + (size_t)b * Nn * 256;
  float acc = 0.f;
  for (int n = 0; n < Nn; ++n) acc += ab[n] * hb[(size_t)n * 256 + d];
  pooled[b * 256 + d] = acc;
}

// out[b,c] = pooled[b] . cls_w[:,c] + cls_b[c]
__global__ __launch_bounds__(128)
void cls_kernel(const float* __restrict__ pooled, const float* __restrict__ cw,
                const float* __restrict__ cb, float* __restrict__ out) {
  const int b = blockIdx.x;
  const float* p = pooled + b * 256;
  for (int c = threadIdx.x; c < 104; c += 128) {
    float acc = cb[c];
    for (int d = 0; d < 256; ++d) acc += p[d] * cw[d * 104 + c];
    out[b * 104 + c] = acc;
  }
}

// ---------------------------------------------------------------------------
// Host orchestration
// ---------------------------------------------------------------------------
extern "C" void kernel_launch(void* const* d_in, const int* in_sizes, int n_in,
                              void* d_out, int out_size, void* d_ws, size_t ws_size,
                              hipStream_t stream) {
  const int Bv = 32, Nn = 4681;
  static const int SZ[5]   = {1, 8, 64, 512, 4096};
  static const int OFFv[6] = {0, 1, 9, 73, 585, 4681};
  const int GC = 2048;    // group chunk (bottom-up): 16384 child rows
  const int RCTD = 16384; // row chunk (top-down)

  const int*   types = (const int*)  d_in[0];
  const float* emb   = (const float*)d_in[1];
  const float* wq  = (const float*)d_in[2],  *bq  = (const float*)d_in[3];
  const float* wk  = (const float*)d_in[4],  *bk  = (const float*)d_in[5];
  const float* wv  = (const float*)d_in[6],  *bv  = (const float*)d_in[7];
  const float* wqp = (const float*)d_in[8],  *bqp = (const float*)d_in[9];
  const float* wkp = (const float*)d_in[10], *bkp = (const float*)d_in[11];
  const float* wvp = (const float*)d_in[12], *bvp = (const float*)d_in[13];
  const float* wo_s = (const float*)d_in[14], *bo_s = (const float*)d_in[15];
  const float* wo_p = (const float*)d_in[16], *bo_p = (const float*)d_in[17];
  const float* w1 = (const float*)d_in[18], *b1 = (const float*)d_in[19];
  const float* w2 = (const float*)d_in[20], *b2 = (const float*)d_in[21];
  const float* ln_g = (const float*)d_in[22], *ln_b = (const float*)d_in[23];
  const float* t_w1 = (const float*)d_in[24], *t_b1 = (const float*)d_in[25];
  const float* t_w2 = (const float*)d_in[26], *t_b2 = (const float*)d_in[27];
  const float* t_ln_g = (const float*)d_in[28], *t_ln_b = (const float*)d_in[29];
  const float* gate_w = (const float*)d_in[30], *gate_b = (const float*)d_in[31];
  const float* cls_w  = (const float*)d_in[32], *cls_b  = (const float*)d_in[33];
  float* out = (float*)d_out;

  // ---- workspace bump allocation ----
  char* wsp = (char*)d_ws;
  auto alloc = [&](size_t bytes) -> void* {
    void* p = (void*)wsp; wsp += (bytes + 255) & ~(size_t)255; return p;
  };
  float* hall = (float*)alloc((size_t)Bv * Nn * 256 * 4);
  float* pe   = (float*)alloc(8 * 256 * 4);
  _Float16* wq_t  = (_Float16*)alloc(256 * 256 * 2);
  _Float16* wk_t  = (_Float16*)alloc(256 * 256 * 2);
  _Float16* wv_t  = (_Float16*)alloc(256 * 256 * 2);
  _Float16* wqp_t = (_Float16*)alloc(256 * 256 * 2);
  _Float16* wkp_t = (_Float16*)alloc(256 * 256 * 2);
  _Float16* wvp_t = (_Float16*)alloc(256 * 256 * 2);
  _Float16* wos_t = (_Float16*)alloc(256 * 256 * 2);
  _Float16* wop_t = (_Float16*)alloc(256 * 256 * 2);
  _Float16* w1_t  = (_Float16*)alloc(256 * 1024 * 2);
  _Float16* w2_t  = (_Float16*)alloc(1024 * 256 * 2);
  _Float16* tw1_t = (_Float16*)alloc(256 * 1024 * 2);
  _Float16* tw2_t = (_Float16*)alloc(1024 * 256 * 2);
  const size_t RBUF = (size_t)16384 * 256 * 4;
  float* xb    = (float*)alloc(RBUF);
  float* qb    = (float*)alloc(RBUF);
  float* kb    = (float*)alloc(RBUF);
  float* vb    = (float*)alloc(RBUF);
  float* attnb = (float*)alloc(RBUF);
  float* x2b   = (float*)alloc(RBUF);
  float* nkb   = (float*)alloc(RBUF);
  float* nvb   = (float*)alloc(RBUF);
  float* parb  = (float*)alloc((size_t)GC * 256 * 4);
  float* pqb   = (float*)alloc((size_t)GC * 256 * 4);
  float* ctxb  = (float*)alloc((size_t)GC * 256 * 4);
  float* ctx2b = (float*)alloc((size_t)GC * 256 * 4);
  float* f1b   = (float*)alloc((size_t)16384 * 1024 * 4);
  float* sbuf  = (float*)alloc((size_t)Bv * Nn * 4);
  float* pooled = (float*)alloc((size_t)Bv * 256 * 4);

  auto gemm = [&](const float* A, const _Float16* Wt, const float* bias,
                  const float* res, float* C, int M, int Ncols, int K, int act) {
    dim3 grid((M + 63) / 64, Ncols / 128);
    wmma_gemm<<<grid, 256, 0, stream>>>(A, Wt, bias, res, C, M, Ncols, K, act);
  };
  auto cvt = [&](const float* W, _Float16* Wt, int K, int Ncols) {
    int tot = K * Ncols;
    cvt_wt<<<(tot + 255) / 256, 256, 0, stream>>>(W, Wt, K, Ncols);
  };
  auto ln_d = [&](const float* in, float* o, const float* g, const float* b, int rows) {
    ln_dense<<<(rows + 7) / 8, 256, 0, stream>>>(in, o, g, b, rows);
  };

  // ---- setup ----
  pe_init<<<8, 256, 0, stream>>>(pe);
  cvt(wq, wq_t, 256, 256);   cvt(wk, wk_t, 256, 256);   cvt(wv, wv_t, 256, 256);
  cvt(wqp, wqp_t, 256, 256); cvt(wkp, wkp_t, 256, 256); cvt(wvp, wvp_t, 256, 256);
  cvt(wo_s, wos_t, 256, 256); cvt(wo_p, wop_t, 256, 256);
  cvt(w1, w1_t, 256, 1024);  cvt(w2, w2_t, 1024, 256);
  cvt(t_w1, tw1_t, 256, 1024); cvt(t_w2, tw2_t, 1024, 256);
  embed_gather<<<Bv * Nn, 256, 0, stream>>>(types, emb, hall);

  // ---- bottom-up ----
  for (int l = 3; l >= 0; --l) {
    const int P = SZ[l];
    const int G = Bv * P;
    for (int g0 = 0; g0 < G; g0 += GC) {
      const int gc = (G - g0 < GC) ? (G - g0) : GC;
      const int rc = gc * 8;
      gather_children_pe<<<rc, 256, 0, stream>>>(hall, pe, xb, OFFv[l + 1], P, g0);
      gemm(xb, wq_t, bq, nullptr, qb, rc, 256, 256, 0);
      gemm(xb, wk_t, bk, nullptr, kb, rc, 256, 256, 0);
      gemm(xb, wv_t, bv, nullptr, vb, rc, 256, 256, 0);
      sib_attn<<<gc, 256, 0, stream>>>(qb, kb, vb, attnb);
      gemm(attnb, wos_t, bo_s, xb, x2b, rc, 256, 256, 0);   // + residual x
      ln_d(x2b, x2b, ln_g, ln_b, rc);
      gemm(x2b, wkp_t, bkp, nullptr, nkb, rc, 256, 256, 0);
      gemm(x2b, wvp_t, bvp, nullptr, nvb, rc, 256, 256, 0);
      gather_rows<<<gc, 256, 0, stream>>>(hall, parb, g0, P, OFFv[l]);
      gemm(parb, wqp_t, bqp, nullptr, pqb, gc, 256, 256, 0);
      par_attn<<<(gc + 7) / 8, 256, 0, stream>>>(pqb, nkb, nvb, ctxb, gc);
      gemm(ctxb, wop_t, bo_p, parb, ctx2b, gc, 256, 256, 0); // + residual par
      ln_d(ctx2b, ctx2b, ln_g, ln_b, gc);
      gemm(ctx2b, w1_t, b1, nullptr, f1b, gc, 1024, 256, 1); // GELU
      gemm(f1b, w2_t, b2, ctx2b, xb, gc, 256, 1024, 0);      // + residual ctx2
      ln_scatter<<<(gc + 7) / 8, 256, 0, stream>>>(xb, hall, ln_g, ln_b,
                                                   gc, g0, P, OFFv[l]);
    }
  }

  // ---- top-down ----
  for (int l = 1; l < 5; ++l) {
    const int S = SZ[l];
    const int Rl = Bv * S;
    for (int r0 = 0; r0 < Rl; r0 += RCTD) {
      const int rc = (Rl - r0 < RCTD) ? (Rl - r0) : RCTD;
      td_combine<<<(rc + 7) / 8, 256, 0, stream>>>(hall, xb, rc, r0, S,
                                                   OFFv[l - 1], OFFv[l],
                                                   t_ln_g, t_ln_b);
      gemm(xb, tw1_t, t_b1, nullptr, f1b, rc, 1024, 256, 1); // GELU
      gemm(f1b, tw2_t, t_b2, xb, qb, rc, 256, 1024, 0);      // + residual ctx
      ln_scatter<<<(rc + 7) / 8, 256, 0, stream>>>(qb, hall, t_ln_g, t_ln_b,
                                                   rc, r0, S, OFFv[l]);
    }
  }

  // ---- pooling + classifier ----
  gate_rows<<<(Bv * Nn + 7) / 8, 256, 0, stream>>>(hall, gate_w, gate_b, sbuf, Bv * Nn);
  softmax_rows<<<Bv, 256, 0, stream>>>(sbuf, Nn);
  pool_rows<<<Bv, 256, 0, stream>>>(sbuf, hall, pooled, Nn);
  cls_kernel<<<Bv, 128, 0, stream>>>(pooled, cls_w, cls_b, out);
  (void)in_sizes; (void)n_in; (void)out_size; (void)ws_size;
}
